// MultiHeadSelfAttention_44538810859781
// MI455X (gfx1250) — matrix operand order repaired and checked by reading, not compiled, not run on hardware
//
#include <hip/hip_runtime.h>

// ---------------------------------------------------------------------------
// MI455X (gfx1250) fused multi-head self-attention
//   - bf16 WMMA (v_wmma_f32_16x16x32_bf16) with f32 accumulation for all GEMMs
//   - flash-attention (online softmax): avoids the 536MB score matrix
//   - double-buffered K/V LDS tiles filled with GLOBAL_LOAD_ASYNC_TO_LDS_B128
//     (ASYNCcnt-tracked, waited with s_wait_asynccnt) so tile loads overlap
//     the 16 WMMAs + softmax of the previous tile
//   - 16x128 output tile per wave in the dense GEMMs (8 accumulators) for
//     2x A-fragment reuse and 8 independent wmma/load chains per k-step
// ---------------------------------------------------------------------------

#define DEV __device__ __forceinline__

typedef __attribute__((ext_vector_type(16))) __bf16         v16bf;
typedef __attribute__((ext_vector_type(8)))  float          v8f;
typedef __attribute__((ext_vector_type(4)))  unsigned int   u32x4;
typedef __attribute__((ext_vector_type(4)))  unsigned short u16x4;
typedef __attribute__((ext_vector_type(4)))  int            v4i;

static constexpr int B_  = 4;
static constexpr int N_  = 2048;
static constexpr int C_  = 512;
static constexpr int H_  = 8;
static constexpr int HD_ = 64;
static constexpr float SCALE_ = 0.125f;   // HD^-0.5

union Frag { u32x4 q[2]; v16bf v; };

#if defined(__has_builtin)
#if __has_builtin(__builtin_amdgcn_global_load_async_to_lds_b128)
#define HAVE_ASYNC_LDS 1
#endif
#endif
#ifndef HAVE_ASYNC_LDS
#define HAVE_ASYNC_LDS 0
#endif

#if HAVE_ASYNC_LDS
// builtin signature (from hipcc diagnostic): (int4 AS1*, int4 AS3*, imm, imm)
typedef __attribute__((address_space(1))) v4i GV4i;
typedef __attribute__((address_space(3))) v4i LV4i;
// one 16B lane-granular async copy, memory -> LDS, tracked by ASYNCcnt
DEV void async_cp16(const unsigned short* g, unsigned short* l) {
  __builtin_amdgcn_global_load_async_to_lds_b128((GV4i*)g, (LV4i*)l, 0, 0);
}
#endif

DEV unsigned short f32_bf16(float f) {
  unsigned int u = __float_as_uint(f);
  u += 0x7fffu + ((u >> 16) & 1u);        // round-to-nearest-even
  return (unsigned short)(u >> 16);
}

// A-matrix fragment (16x32 bf16, MxK): lane holds row=lane&15; VGPR0-7 hold
// K = kb..kb+15 (kb = k0 + (lane>=16 ? 16:0)).
DEV v16bf frag_a(const unsigned short* base, int ld, int row, int k0, int lane) {
  int kb = k0 + ((lane & 16) ? 16 : 0);
  const unsigned short* r = base + (size_t)row * ld + kb;
  Frag f;
  f.q[0] = *(const u32x4*)r;
  f.q[1] = *(const u32x4*)(r + 8);
  return f.v;
}

// B-matrix fragment (32x16 bf16, KxN) from Bt stored row-major by N (Bt[n][k]):
// lane holds col=lane&15; VGPR j holds K pair kb+2j (kb = k0 + (lane>=16?16:0)).
DEV v16bf frag_b(const unsigned short* base, int ld, int ncol, int k0, int lane) {
  int kb = k0 + ((lane & 16) ? 16 : 0);
  const unsigned short* r = base + (size_t)ncol * ld + kb;
  Frag f;
  f.q[0] = *(const u32x4*)r;
  f.q[1] = *(const u32x4*)(r + 8);
  return f.v;
}

// ---------------------------------------------------------------------------
__global__ __launch_bounds__(256) void cast_bf16_kernel(
    const float* __restrict__ src, unsigned short* __restrict__ dst, int n) {
  int i = (blockIdx.x * 256 + threadIdx.x) * 4;
  if (i < n) {
    float4 f = *(const float4*)(src + i);
    u16x4 o;
    o.x = f32_bf16(f.x); o.y = f32_bf16(f.y);
    o.z = f32_bf16(f.z); o.w = f32_bf16(f.w);
    *(u16x4*)(dst + i) = o;
  }
}

// ---------------------------------------------------------------------------
// QKV = Xb[8192,512] @ Wqkv^T[512,1536] + b.  Epilogue scatters into
// Q [b,h,n,hd] (pre-scaled), K [b,h,n,hd], Vt [b,h,hd,n] as bf16.
// Block = 8 waves, each wave owns a 16x128 output tile (8 accumulators).
__global__ __launch_bounds__(256) void qkv_gemm_kernel(
    const unsigned short* __restrict__ Xb, const unsigned short* __restrict__ Wb,
    const float* __restrict__ bias,
    unsigned short* __restrict__ Qb, unsigned short* __restrict__ Kb,
    unsigned short* __restrict__ Vtb) {
  int lane = threadIdx.x & 31, w = threadIdx.x >> 5;
  int m0 = blockIdx.y * 128 + w * 16;
  int n0 = blockIdx.x * 128;
  int arow = m0 + (lane & 15);

  v8f acc[8] = {};
  for (int k0 = 0; k0 < C_; k0 += 32) {
    v16bf a = frag_a(Xb, C_, arow, k0, lane);
#pragma unroll
    for (int t = 0; t < 8; ++t) {
      v16bf b = frag_b(Wb, C_, n0 + t * 16 + (lane & 15), k0, lane);
      acc[t] = __builtin_amdgcn_wmma_f32_16x16x32_bf16(
          false, a, false, b, (short)0, acc[t], false, false);
    }
  }

  int hi = (lane & 16) ? 8 : 0;
  int ncl = lane & 15;
#pragma unroll
  for (int t = 0; t < 8; ++t) {
    int n = n0 + t * 16 + ncl;
    float bv = bias[n];
    int chunk = n >> 9, c = n & 511, h = c >> 6, hd = c & 63;
#pragma unroll
    for (int j = 0; j < 8; ++j) {
      int m = m0 + j + hi;
      int b = m >> 11, ns = m & 2047;
      float v = acc[t][j] + bv;
      if (chunk == 0)
        Qb[(((size_t)(b * H_ + h)) * N_ + ns) * HD_ + hd] = f32_bf16(v * SCALE_);
      else if (chunk == 1)
        Kb[(((size_t)(b * H_ + h)) * N_ + ns) * HD_ + hd] = f32_bf16(v);
      else
        Vtb[(((size_t)(b * H_ + h)) * HD_ + hd) * N_ + ns] = f32_bf16(v);
    }
  }
}

// ---------------------------------------------------------------------------
// Cooperative 64x64 K/V tile prefetch into LDS (256 threads).
// Async path: each lane issues 16B GLOBAL_LOAD_ASYNC_TO_LDS_B128 copies.
DEV void tile_prefetch(const unsigned short* Ksrc,   // K + kb*HD_ (contiguous)
                       const unsigned short* Vsrc,   // Vt + kb    (row stride N_)
                       unsigned short* ktb, unsigned short* vtb, int tid) {
#if HAVE_ASYNC_LDS
  async_cp16(Ksrc + tid * 8,        ktb + tid * 8);
  async_cp16(Ksrc + 2048 + tid * 8, ktb + 2048 + tid * 8);
  int r = tid >> 2, seg = tid & 3;
  const unsigned short* vr = Vsrc + (size_t)r * N_ + seg * 16;
  unsigned short* vd = vtb + r * 64 + seg * 16;
  async_cp16(vr, vd);
  async_cp16(vr + 8, vd + 8);
#else
  const u32x4* ks = (const u32x4*)Ksrc;
  u32x4* kd = (u32x4*)ktb;
  kd[tid]       = ks[tid];
  kd[tid + 256] = ks[tid + 256];
  int r = tid >> 2, seg = tid & 3;
  const u32x4* vs = (const u32x4*)(Vsrc + (size_t)r * N_);
  u32x4* vd = (u32x4*)(vtb + r * 64);
  vd[seg * 2]     = vs[seg * 2];
  vd[seg * 2 + 1] = vs[seg * 2 + 1];
#endif
}

DEV void tile_wait() {
#if HAVE_ASYNC_LDS
  asm volatile("s_wait_asynccnt 0" ::: "memory");
#endif
}

// ---------------------------------------------------------------------------
// Flash attention: grid = (N/128, B*H).  8 waves x 16 query rows each.
// Double-buffered K/V LDS tiles; async prefetch of tile i+1 overlaps the
// 16 WMMAs + online softmax of tile i.  Q fragments hoisted out of the loop.
__global__ __launch_bounds__(256) void flash_attn_kernel(
    const unsigned short* __restrict__ Qb, const unsigned short* __restrict__ Kb,
    const unsigned short* __restrict__ Vtb, unsigned short* __restrict__ Ctx) {
  __shared__ unsigned short kt[2][64 * 64];     // [key][hd]
  __shared__ unsigned short vt[2][64 * 64];     // [hd][key]
  __shared__ unsigned short pb[8][16 * 64];     // per-wave P tile, [row][key]

  int tid = threadIdx.x;
  int lane = tid & 31, w = tid >> 5;
  int bh = blockIdx.y;
  const unsigned short* Q  = Qb  + (size_t)bh * N_ * HD_;
  const unsigned short* K  = Kb  + (size_t)bh * N_ * HD_;
  const unsigned short* Vt = Vtb + (size_t)bh * HD_ * N_;
  int q0 = blockIdx.x * 128 + w * 16;

  int arow = q0 + (lane & 15);
  v16bf aq0 = frag_a(Q, HD_, arow, 0, lane);    // invariant over key loop
  v16bf aq1 = frag_a(Q, HD_, arow, 32, lane);

  v8f o[4] = {};
  float mR[8], lR[8];
#pragma unroll
  for (int j = 0; j < 8; ++j) { mR[j] = -1e30f; lR[j] = 0.f; }

  int hi = (lane & 16) ? 8 : 0;
  int ncl = lane & 15;

  tile_prefetch(K, Vt, kt[0], vt[0], tid);

  for (int it = 0; it < N_ / 64; ++it) {
    int cur = it & 1;
    tile_wait();           // my async copies (tile `it`) have landed in LDS
    __syncthreads();       // everyone's copies landed; prev iter fully done

    if (it + 1 < N_ / 64)  // prefetch next tile into the other buffer
      tile_prefetch(K + (size_t)(it + 1) * 64 * HD_, Vt + (it + 1) * 64,
                    kt[cur ^ 1], vt[cur ^ 1], tid);

    // S = Q @ K^T   (16 x 64 in four 16x16 C-fragments)
    v8f s[4] = {};
#pragma unroll
    for (int t = 0; t < 4; ++t) {
      int key = t * 16 + ncl;
      int db = (lane & 16) ? 16 : 0;
      Frag b0, b1;
      b0.q[0] = *(const u32x4*)&kt[cur][key * 64 + db];
      b0.q[1] = *(const u32x4*)&kt[cur][key * 64 + db + 8];
      b1.q[0] = *(const u32x4*)&kt[cur][key * 64 + 32 + db];
      b1.q[1] = *(const u32x4*)&kt[cur][key * 64 + 32 + db + 8];
      s[t] = __builtin_amdgcn_wmma_f32_16x16x32_bf16(
          false, aq0, false, b0.v, (short)0, s[t], false, false);
      s[t] = __builtin_amdgcn_wmma_f32_16x16x32_bf16(
          false, aq1, false, b1.v, (short)0, s[t], false, false);
    }

    // online softmax: C-frag row j lives on VGPR j across 16 lanes of a half
#pragma unroll
    for (int j = 0; j < 8; ++j) {
      float mt = fmaxf(fmaxf(s[0][j], s[1][j]), fmaxf(s[2][j], s[3][j]));
#pragma unroll
      for (int d = 8; d >= 1; d >>= 1) mt = fmaxf(mt, __shfl_xor(mt, d, 32));
      float mnew = fmaxf(mR[j], mt);
      float corr = __expf(mR[j] - mnew);
      mR[j] = mnew;
      lR[j] *= corr;
#pragma unroll
      for (int t = 0; t < 4; ++t) o[t][j] *= corr;
      float rs = 0.f;
#pragma unroll
      for (int t = 0; t < 4; ++t) {
        float p = __expf(s[t][j] - mnew);
        s[t][j] = p;
        rs += p;
      }
#pragma unroll
      for (int d = 8; d >= 1; d >>= 1) rs += __shfl_xor(rs, d, 32);
      lR[j] += rs;
    }

    // C-layout -> A-layout for P via per-wave LDS tile
#pragma unroll
    for (int t = 0; t < 4; ++t)
#pragma unroll
      for (int j = 0; j < 8; ++j)
        pb[w][(j + hi) * 64 + t * 16 + ncl] = f32_bf16(s[t][j]);
    asm volatile("s_wait_dscnt 0" ::: "memory");  // same-wave LDS RAW

    // O += P @ V
#pragma unroll
    for (int ck = 0; ck < 2; ++ck) {
      int ab = (lane & 16) ? 16 : 0;
      Frag ap;
      ap.q[0] = *(const u32x4*)&pb[w][(lane & 15) * 64 + ck * 32 + ab];
      ap.q[1] = *(const u32x4*)&pb[w][(lane & 15) * 64 + ck * 32 + 8 + ab];
#pragma unroll
      for (int t = 0; t < 4; ++t) {
        int hd = t * 16 + ncl;
        int kbv = (lane & 16) ? 16 : 0;
        Frag bv;
        bv.q[0] = *(const u32x4*)&vt[cur][hd * 64 + ck * 32 + kbv];
        bv.q[1] = *(const u32x4*)&vt[cur][hd * 64 + ck * 32 + kbv + 8];
        o[t] = __builtin_amdgcn_wmma_f32_16x16x32_bf16(
            false, ap.v, false, bv.v, (short)0, o[t], false, false);
      }
    }
  }

  // normalize + scatter to ctx [B,N,H*HD] (fuses the 0,2,1,3 transpose)
  int b = bh >> 3, h = bh & 7;
#pragma unroll
  for (int j = 0; j < 8; ++j) {
    float inv = 1.f / lR[j];
    int q = q0 + j + hi;
#pragma unroll
    for (int t = 0; t < 4; ++t) {
      int c = h * HD_ + t * 16 + ncl;
      Ctx[((size_t)(b * N_ + q)) * C_ + c] = f32_bf16(o[t][j] * inv);
    }
  }
}

// ---------------------------------------------------------------------------
// out = Ctx[8192,512] @ Wproj^T[512,512] + b_proj  (fp32 output)
// Wave tile 16x128 (8 accumulators).
__global__ __launch_bounds__(256) void proj_gemm_kernel(
    const unsigned short* __restrict__ Ab, const unsigned short* __restrict__ Wb,
    const float* __restrict__ bias, float* __restrict__ out) {
  int lane = threadIdx.x & 31, w = threadIdx.x >> 5;
  int m0 = blockIdx.y * 128 + w * 16;
  int n0 = blockIdx.x * 128;
  int arow = m0 + (lane & 15);

  v8f acc[8] = {};
  for (int k0 = 0; k0 < C_; k0 += 32) {
    v16bf a = frag_a(Ab, C_, arow, k0, lane);
#pragma unroll
    for (int t = 0; t < 8; ++t) {
      v16bf b = frag_b(Wb, C_, n0 + t * 16 + (lane & 15), k0, lane);
      acc[t] = __builtin_amdgcn_wmma_f32_16x16x32_bf16(
          false, a, false, b, (short)0, acc[t], false, false);
    }
  }

  int hi = (lane & 16) ? 8 : 0;
  int ncl = lane & 15;
#pragma unroll
  for (int t = 0; t < 8; ++t) {
    int n = n0 + t * 16 + ncl;
    float bv = bias[n];
#pragma unroll
    for (int j = 0; j < 8; ++j) {
      int m = m0 + j + hi;
      out[(size_t)m * C_ + n] = acc[t][j] + bv;
    }
  }
}

// ---------------------------------------------------------------------------
extern "C" void kernel_launch(void* const* d_in, const int* in_sizes, int n_in,
                              void* d_out, int out_size, void* d_ws, size_t ws_size,
                              hipStream_t stream) {
  const float* x      = (const float*)d_in[0];
  const float* w_qkv  = (const float*)d_in[1];
  const float* b_qkv  = (const float*)d_in[2];
  const float* w_proj = (const float*)d_in[3];
  const float* b_proj = (const float*)d_in[4];
  float* out = (float*)d_out;

  char* ws = (char*)d_ws;
  size_t off = 0;
  auto alloc = [&](size_t bytes) {
    char* p = ws + off;
    off += (bytes + 255) & ~(size_t)255;
    return p;
  };

  const size_t n_x  = (size_t)B_ * N_ * C_;       // 4,194,304
  const size_t n_wq = (size_t)3 * C_ * C_;        // 786,432
  const size_t n_wp = (size_t)C_ * C_;            // 262,144
  const size_t n_bhnd = (size_t)B_ * H_ * N_ * HD_;

  unsigned short* Xb  = (unsigned short*)alloc(n_x * 2);
  unsigned short* Wqb = (unsigned short*)alloc(n_wq * 2);
  unsigned short* Wpb = (unsigned short*)alloc(n_wp * 2);
  unsigned short* Qb  = (unsigned short*)alloc(n_bhnd * 2);
  unsigned short* Kb  = (unsigned short*)alloc(n_bhnd * 2);
  unsigned short* Vtb = (unsigned short*)alloc(n_bhnd * 2);
  unsigned short* Ctx = (unsigned short*)alloc(n_x * 2);

  // 1) fp32 -> bf16 casts
  cast_bf16_kernel<<<dim3((unsigned)((n_x / 4 + 255) / 256)), 256, 0, stream>>>(
      x, Xb, (int)n_x);
  cast_bf16_kernel<<<dim3((unsigned)((n_wq / 4 + 255) / 256)), 256, 0, stream>>>(
      w_qkv, Wqb, (int)n_wq);
  cast_bf16_kernel<<<dim3((unsigned)((n_wp / 4 + 255) / 256)), 256, 0, stream>>>(
      w_proj, Wpb, (int)n_wp);

  // 2) QKV projection (WMMA), scatter into Q (scaled), K, V^T
  qkv_gemm_kernel<<<dim3(3 * C_ / 128, B_ * N_ / 128), 256, 0, stream>>>(
      Xb, Wqb, b_qkv, Qb, Kb, Vtb);

  // 3) flash attention (WMMA + async double-buffered LDS tiles)
  flash_attn_kernel<<<dim3(N_ / 128, B_ * H_), 256, 0, stream>>>(Qb, Kb, Vtb, Ctx);

  // 4) output projection (WMMA) + bias -> fp32 d_out
  proj_gemm_kernel<<<dim3(C_ / 128, B_ * N_ / 128), 256, 0, stream>>>(
      Ctx, Wpb, b_proj, out);
}
